// MSDeformableAttention_27504970563832
// MI455X (gfx1250) — matrix operand order repaired and checked by reading, not compiled, not run on hardware
//
#include <hip/hip_runtime.h>
#include <hip/hip_bf16.h>

// ---------------- problem constants ----------------
#define D_MODEL   256
#define N_HEADS   8
#define N_LEVELS  4
#define N_POINTS  4
#define HEAD_DIM  32
#define LEN_Q     13294
#define LEN_V     13294
#define BSZ       2
#define M_ROWS    (BSZ * LEN_Q)          // 26588
#define K_DIM     256

typedef __attribute__((ext_vector_type(16))) _Float16 v16h;
typedef __attribute__((ext_vector_type(8)))  _Float16 v8h;
typedef __attribute__((ext_vector_type(8)))  float    v8f;

union HFrag { v16h v; _Float16 e[16]; v8h h2[2]; };
union F8    { v8f  v; float    f[8]; };

static __device__ __forceinline__ int imin(int a, int b) { return a < b ? a : b; }

// ---------------- pack [W0 | W1] (K x N0, K x N1 fp32 row-major) into WMMA B-fragment layout ---
// Packed layout: Wpk[kt][ct][lane][16 halves], kt = K/32 tiles, ct = (N0+N1)/16 tiles.
// lane L (n16=L&15, half=L>>4), element i (v=i>>1, odd=i&1):
//   k = kt*32 + half*16 + 2*v + odd ; n = ct*16 + n16
__global__ void pack_w_kernel(const float* __restrict__ W0, int N0,
                              const float* __restrict__ W1, int N1,
                              _Float16* __restrict__ Wpk) {
    const int N   = N0 + N1;
    const int idx = blockIdx.x * blockDim.x + threadIdx.x;
    if (idx >= K_DIM * N) return;
    const int i    = idx & 15;
    const int lane = (idx >> 4) & 31;
    const int rest = idx >> 9;           // kt*CT + ct
    const int CT   = N >> 4;
    const int kt   = rest / CT;
    const int ct   = rest - kt * CT;
    const int k = kt * 32 + (lane >> 4) * 16 + (i & ~1) + (i & 1);
    const int n = ct * 16 + (lane & 15);
    const float w = (n < N0) ? W0[(size_t)k * N0 + n] : W1[(size_t)k * N1 + (n - N0)];
    Wpk[idx] = (_Float16)w;
}

// ---------------- concat two fp32 vectors ----------------
__global__ void concat2_f32(const float* __restrict__ a, int na,
                            const float* __restrict__ b, int nb,
                            float* __restrict__ out) {
    const int i = blockIdx.x * blockDim.x + threadIdx.x;
    if (i < na) out[i] = a[i];
    else if (i < na + nb) out[i] = b[i - na];
}

// ---------------- A fragment loader ----------------
template <bool AF32>
static __device__ __forceinline__ void load_a_frag(HFrag& a, const void* arowp, int kt, int half) {
    if constexpr (AF32) {   // fp32 source, convert in-register (no f16 staging pass)
        const float* ap = (const float*)arowp + kt * 32 + half * 16;
        F8 lo, hi;
        lo.v = *(const v8f*)(ap);
        hi.v = *(const v8f*)(ap + 8);
        #pragma unroll
        for (int j = 0; j < 8; ++j) {
            a.e[j]     = (_Float16)lo.f[j];
            a.e[8 + j] = (_Float16)hi.f[j];
        }
    } else {
        const _Float16* ap = (const _Float16*)arowp + kt * 32 + half * 16;
        a.h2[0] = *(const v8h*)(ap);
        a.h2[1] = *(const v8h*)(ap + 8);
    }
}

// ---------------- WMMA GEMM: C[M x N] = A[M x K] * W[K x N] + bias ----------------
// NT 16-col tiles per wave; block = 8 waves (4 row-tiles x 2 col-groups) = 64 x (32*NT) of C.
// B fragments for the whole block column (all K) are staged in LDS once (64 KB),
// so the inner loop is: global A (double-buffered) + ds_load B + WMMA.
// MODE 0: fp32 store out0[M x N]               (out-projection -> d_out)
// MODE 1: f16 scatter outH[b][head][LEN_V][32] (value projection)
// MODE 2: split store: col<256 -> out0[M x 256], col>=256 -> out1[M x 128]  (off+attn fused)
template <int NT, bool AF32, int MODE>
__global__ __launch_bounds__(256) void gemm_wmma_kernel(
    const void*     __restrict__ A,      // M x K row-major (fp32 if AF32 else f16)
    const _Float16* __restrict__ Wpk,    // packed B fragments
    const float*    __restrict__ bias,   // N
    float*          __restrict__ out0,
    float*          __restrict__ out1,
    _Float16*       __restrict__ outH,
    int M, int N)
{
    constexpr int KT    = K_DIM / 32;        // 8 k-steps
    constexpr int TILES = 2 * NT;            // col tiles per block (8)
    constexpr int FRAG  = 32 * 16;           // halves per fragment (one tile, one k-step)

    __shared__ _Float16 smem[KT * TILES * FRAG];   // 8*8*512 halves = 64 KB

    const int lane = threadIdx.x & 31;
    const int wid  = threadIdx.x >> 5;
    const int n16  = lane & 15;
    const int half = lane >> 4;

    const int rm       = blockIdx.x * 64 + (wid >> 1) * 16;
    const int tileBase = blockIdx.y * TILES;           // first col tile of block
    const int ctl      = (wid & 1) * NT;               // this wave's first local tile
    const int CT       = N >> 4;

    // ---- cooperative stage of packed B fragments into LDS ----
    {
        constexpr int CHUNKS = (TILES * FRAG) / (256 * 8);   // v8h chunks per thread per kt (=2)
        #pragma unroll
        for (int kt = 0; kt < KT; ++kt) {
            const _Float16* src = Wpk + ((size_t)kt * CT + tileBase) * FRAG;
            _Float16*       dst = smem + kt * TILES * FRAG;
            #pragma unroll
            for (int j = 0; j < CHUNKS; ++j) {
                const int e = (j * 256 + (int)threadIdx.x) * 8;
                *(v8h*)(dst + e) = *(const v8h*)(src + e);
            }
        }
    }
    __syncthreads();

    const int arow = imin(rm + n16, M - 1);      // clamp: EXEC stays all-1 for WMMA
    const void* arowp = (const char*)A + (size_t)arow * K_DIM * (AF32 ? 4 : 2);

    v8f acc[NT] = {};

    // A double-buffered in registers; B read from LDS just-in-time
    HFrag a_c, a_n;
    load_a_frag<AF32>(a_c, arowp, 0, half);

    #pragma unroll
    for (int kt = 0; kt < KT; ++kt) {
        if (kt + 1 < KT) load_a_frag<AF32>(a_n, arowp, kt + 1, half);
        #pragma unroll
        for (int c = 0; c < NT; ++c) {
            HFrag b;
            const v8h* bp = (const v8h*)(smem + ((kt * TILES + ctl + c) * 32 + lane) * 16);
            b.h2[0] = bp[0];
            b.h2[1] = bp[1];
            acc[c] = __builtin_amdgcn_wmma_f32_16x16x32_f16(
                false, a_c.v, false, b.v, (short)0, acc[c], false, false);
        }
        if (kt + 1 < KT) a_c = a_n;
    }

    float bv[NT];
    #pragma unroll
    for (int c = 0; c < NT; ++c) bv[c] = bias[(tileBase + ctl + c) * 16 + n16];

    #pragma unroll
    for (int r = 0; r < 8; ++r) {
        const int row = rm + half * 8 + r;
        if (row >= M) continue;
        int bb = 0, ii = 0;
        if constexpr (MODE == 1) { bb = row / LEN_V; ii = row - bb * LEN_V; }
        #pragma unroll
        for (int c = 0; c < NT; ++c) {
            const int col = (tileBase + ctl + c) * 16 + n16;
            const float vv = acc[c][r] + bv[c];
            if constexpr (MODE == 0) {
                out0[(size_t)row * N + col] = vv;
            } else if constexpr (MODE == 1) {
                const int h = col >> 5, d = col & 31;
                outH[((size_t)(bb * N_HEADS + h) * LEN_V + ii) * HEAD_DIM + d] = (_Float16)vv;
            } else {
                if (col < 256) out0[(size_t)row * 256 + col] = vv;
                else           out1[(size_t)row * 128 + (col - 256)] = vv;
            }
        }
    }
}

// ---------------- softmax over 16 attention logits, in place ----------------
__global__ void softmax16_kernel(float* __restrict__ attn, int rows) {
    const int i = blockIdx.x * blockDim.x + threadIdx.x;
    if (i >= rows) return;
    float* p = attn + (size_t)i * 16;
    float m = p[0];
    #pragma unroll
    for (int j = 1; j < 16; ++j) m = fmaxf(m, p[j]);
    float e[16]; float s = 0.f;
    #pragma unroll
    for (int j = 0; j < 16; ++j) { e[j] = __expf(p[j] - m); s += e[j]; }
    const float inv = 1.f / s;
    #pragma unroll
    for (int j = 0; j < 16; ++j) p[j] = e[j] * inv;
}

// ---------------- bilinear gather + attention-weighted sum ----------------
// one wave per (b, q, head); lane = channel (HEAD_DIM == warpSize == 32)
__global__ __launch_bounds__(256) void sample_kernel(
    const float*    __restrict__ refp,   // [B][LEN_Q][N_LEVELS][2]
    const float*    __restrict__ off,    // [M][256]
    const float*    __restrict__ attn,   // [M][128], post-softmax
    const _Float16* __restrict__ valh,   // [B][N_HEADS][LEN_V][HEAD_DIM]
    _Float16*       __restrict__ ctxh)   // [M][256]
{
    const int lane = threadIdx.x & 31;
    const int gw   = blockIdx.x * 8 + (threadIdx.x >> 5);
    const int h    = gw & 7;
    const int row  = gw >> 3;            // b * LEN_Q + q  (grid sized exactly)
    const int b    = row / LEN_Q;

    const float* __restrict__ offRow  = off  + (size_t)row * 256 + h * 32;
    const float* __restrict__ attnRow = attn + (size_t)row * 128 + h * 16;
    const float* __restrict__ refRow  = refp + (size_t)row * (N_LEVELS * 2);
    const _Float16* __restrict__ vbase =
        valh + (size_t)(b * N_HEADS + h) * LEN_V * HEAD_DIM + lane;

    __builtin_prefetch(offRow, 0, 0);    // global_prefetch_b8: start the dependent rows early
    __builtin_prefetch(attnRow, 0, 0);
    __builtin_prefetch(refRow, 0, 0);

    float acc = 0.f;
    #pragma unroll
    for (int l = 0; l < N_LEVELS; ++l) {
        const int HH = (l == 0) ? 100 : (l == 1) ? 50 : (l == 2) ? 25 : 13;
        const int WW = HH;
        const int start = (l == 0) ? 0 : (l == 1) ? 10000 : (l == 2) ? 12500 : 13125;
        const float fw = (float)WW, fh = (float)HH;
        const float rx = refRow[l * 2 + 0];
        const float ry = refRow[l * 2 + 1];
        const _Float16* __restrict__ vlev = vbase + (size_t)start * HEAD_DIM;

        #pragma unroll
        for (int p = 0; p < N_POINTS; ++p) {
            const float ox = offRow[(l * 4 + p) * 2 + 0];
            const float oy = offRow[(l * 4 + p) * 2 + 1];
            const float aw = attnRow[l * 4 + p];

            const float x = (rx + ox / fw) * fw - 0.5f;
            const float y = (ry + oy / fh) * fh - 0.5f;
            const float x0f = floorf(x), y0f = floorf(y);
            const int   x0 = (int)x0f,  y0 = (int)y0f;
            const float wx1 = x - x0f,  wy1 = y - y0f;
            const float wx0 = 1.f - wx1, wy0 = 1.f - wy1;

            #pragma unroll
            for (int cy = 0; cy < 2; ++cy) {
                #pragma unroll
                for (int cx = 0; cx < 2; ++cx) {
                    const int ix = x0 + cx;
                    const int iy = y0 + cy;
                    if (ix >= 0 && ix < WW && iy >= 0 && iy < HH) {
                        const float wt = (cx ? wx1 : wx0) * (cy ? wy1 : wy0);
                        const float g  = (float)vlev[(size_t)(iy * WW + ix) * HEAD_DIM];
                        acc += aw * wt * g;
                    }
                }
            }
        }
    }
    ctxh[(size_t)row * 256 + h * 32 + lane] = (_Float16)acc;
}

// ---------------- host-side launch ----------------
extern "C" void kernel_launch(void* const* d_in, const int* in_sizes, int n_in,
                              void* d_out, int out_size, void* d_ws, size_t ws_size,
                              hipStream_t stream) {
    (void)in_sizes; (void)n_in; (void)out_size; (void)ws_size;

    const float* query  = (const float*)d_in[0];
    const float* refp   = (const float*)d_in[1];
    const float* value  = (const float*)d_in[2];
    /* d_in[3] spatial_shapes: compile-time constants */
    const float* W_off  = (const float*)d_in[4];
    const float* b_off  = (const float*)d_in[5];
    const float* W_attn = (const float*)d_in[6];
    const float* b_attn = (const float*)d_in[7];
    const float* W_val  = (const float*)d_in[8];
    const float* b_val  = (const float*)d_in[9];
    const float* W_out  = (const float*)d_in[10];
    const float* b_out  = (const float*)d_in[11];
    float* out = (float*)d_out;

    char* ws = (char*)d_ws;
    size_t cur = 0;
    auto carve = [&](size_t bytes) -> char* {
        char* p = ws + cur;
        cur += (bytes + 255) & ~(size_t)255;
        return p;
    };

    const size_t NQ = (size_t)M_ROWS * 256;       // 6,806,528
    _Float16* valh   = (_Float16*)carve(NQ * 2);                  // projected val [b][h][LEN_V][32]
    _Float16* ctxh   = (_Float16*)carve(NQ * 2);                  // sampled context f16
    _Float16* Wvpk   = (_Float16*)carve((size_t)256 * 256 * 2);   // packed B fragments
    _Float16* Wfpk   = (_Float16*)carve((size_t)256 * 384 * 2);   // W_off || W_attn packed
    _Float16* Woutpk = (_Float16*)carve((size_t)256 * 256 * 2);
    float*    biasF  = (float*)carve((size_t)384 * 4);            // b_off || b_attn
    float*    offBuf = (float*)carve(NQ * 4);                     // sampling offsets f32
    float*    attnBuf= (float*)carve((size_t)M_ROWS * 128 * 4);   // attention weights f32

    // 1) weight packing (fused fp32->f16 + B-fragment layout) + bias concat
    pack_w_kernel<<<(256 * 256 + 255) / 256, 256, 0, stream>>>(W_val, 256, nullptr, 0, Wvpk);
    pack_w_kernel<<<(256 * 384 + 255) / 256, 256, 0, stream>>>(W_off, 256, W_attn, 128, Wfpk);
    pack_w_kernel<<<(256 * 256 + 255) / 256, 256, 0, stream>>>(W_out, 256, nullptr, 0, Woutpk);
    concat2_f32<<<2, 256, 0, stream>>>(b_off, 256, b_attn, 128, biasF);

    const int mtiles = (M_ROWS + 63) / 64;   // 416

    // 2) val projection (N=256), fp32 A converted in-register, f16 transposed store
    gemm_wmma_kernel<4, true, 1><<<dim3(mtiles, 2), 256, 0, stream>>>(
        value, Wvpk, b_val, nullptr, nullptr, valh, M_ROWS, 256);

    // 3) fused offset + attention projection (N=384), split f32 stores
    gemm_wmma_kernel<4, true, 2><<<dim3(mtiles, 3), 256, 0, stream>>>(
        query, Wfpk, biasF, offBuf, attnBuf, nullptr, M_ROWS, 384);

    // 4) softmax over 16 per (row, head), in place
    {
        const int rows = M_ROWS * N_HEADS;
        softmax16_kernel<<<(rows + 255) / 256, 256, 0, stream>>>(attnBuf, rows);
    }

    // 5) bilinear gather + weighted sum -> ctxh (one wave per (b,q,head))
    sample_kernel<<<M_ROWS, 256, 0, stream>>>(refp, offBuf, attnBuf, valh, ctxh);

    // 6) output projection (N=256) -> d_out fp32 (A already f16)
    gemm_wmma_kernel<4, false, 0><<<dim3(mtiles, 2), 256, 0, stream>>>(
        ctxh, Woutpk, b_out, out, nullptr, nullptr, M_ROWS, 256);
}
